// MultiHeadAttention_300647710879
// MI455X (gfx1250) — compile-verified, matrix operand order checked by reading, not run on hardware
//
#include <hip/hip_runtime.h>
#include <hip/hip_bf16.h>

#define DMODEL 1024
#define DHEAD 64
#define NHEADS 16
#define BATCH 2
#define SEQ 2048
#define MTOT (BATCH * SEQ) /* 4096 */
#define NHB (BATCH * NHEADS) /* 32 */

typedef __bf16 bf16;
typedef bf16 v16bf __attribute__((ext_vector_type(16)));
typedef bf16 v8bf __attribute__((ext_vector_type(8)));
typedef float v8f __attribute__((ext_vector_type(8)));
typedef float v8fv __attribute__((ext_vector_type(8)));
typedef unsigned int v4u __attribute__((ext_vector_type(4)));
typedef int v4i __attribute__((ext_vector_type(4)));
typedef int v8i __attribute__((ext_vector_type(8)));

#if __has_builtin(__builtin_amdgcn_tensor_load_to_lds) && \
    __has_builtin(__builtin_amdgcn_s_wait_tensorcnt)
#define USE_TDM 1
#else
#define USE_TDM 0
#endif

// A 16x32 (or B 32x16) bf16 fragment: per lane (half h = lane>>4), the 16
// elements are two contiguous K-runs: [h*8, h*8+8) and [16+h*8, 16+h*8+8).
// With K contiguous in memory this is two b128 loads (global or ds).
__device__ __forceinline__ v16bf load_frag_bf16(const bf16* rowk0, int h) {
    v8bf lo = *(const v8bf*)(rowk0 + h * 8);
    v8bf hi = *(const v8bf*)(rowk0 + 16 + h * 8);
    return __builtin_shufflevector(lo, hi, 0, 1, 2, 3, 4, 5, 6, 7, 8, 9, 10, 11, 12, 13, 14, 15);
}

// Same, but source is fp32 (converted on the fly -> v_cvt_pk_bf16_f32).
__device__ __forceinline__ v16bf load_frag_f32(const float* __restrict__ rowk0, int h) {
    v8fv lo = *(const v8fv*)(rowk0 + h * 8);
    v8fv hi = *(const v8fv*)(rowk0 + 16 + h * 8);
    v8bf lob = __builtin_convertvector(lo, v8bf);
    v8bf hib = __builtin_convertvector(hi, v8bf);
    return __builtin_shufflevector(lob, hib, 0, 1, 2, 3, 4, 5, 6, 7, 8, 9, 10, 11, 12, 13, 14, 15);
}

#define WMMA_BF16(a, b, c) \
    __builtin_amdgcn_wmma_f32_16x16x32_bf16(false, (a), false, (b), (short)0, (c), false, false)

// ---------------- weight transpose + cvt: src[K,N] f32 -> dst[N,K] bf16 ----------------
__global__ void transpose_cvt(const float* __restrict__ src, bf16* __restrict__ dst) {
    __shared__ bf16 tile[32][33];
    int bx = blockIdx.x % (DMODEL / 32);
    int by = blockIdx.x / (DMODEL / 32);
    int x = bx * 32 + threadIdx.x;
#pragma unroll
    for (int i = threadIdx.y; i < 32; i += 8)
        tile[i][threadIdx.x] = (bf16)src[(size_t)(by * 32 + i) * DMODEL + x];
    __syncthreads();
    int xo = by * 32 + threadIdx.x;
#pragma unroll
    for (int i = threadIdx.y; i < 32; i += 8)
        dst[(size_t)(bx * 32 + i) * DMODEL + xo] = tile[threadIdx.x][i];
}

// ---------------- QKV projection: X(f32)[MTOT,K] @ WT(bf16)[N,K]^T + bias ----------------
__global__ void proj_gemm(const float* __restrict__ X, const bf16* __restrict__ WT,
                          const float* __restrict__ bias, bf16* __restrict__ outp,
                          int transposed_out) {
    int wid = blockIdx.x * (blockDim.x >> 5) + (threadIdx.x >> 5);
    int lane = threadIdx.x & 31;
    const int tiles_n = DMODEL / 64; // 16
    int tm = wid / tiles_n, tn = wid % tiles_n;
    if (tm >= MTOT / 16) return;
    int row0 = tm * 16, col0 = tn * 64;
    int m = lane & 15, h = lane >> 4;

    const float* arow = X + (size_t)(row0 + m) * DMODEL;
    v8f acc[4] = {{}, {}, {}, {}};
    for (int k0 = 0; k0 < DMODEL; k0 += 32) {
        __builtin_prefetch(arow + k0 + 64, 0, 1);
        v16bf a = load_frag_f32(arow + k0, h);
#pragma unroll
        for (int j = 0; j < 4; ++j) {
            v16bf b = load_frag_bf16(WT + (size_t)(col0 + j * 16 + m) * DMODEL + k0, h);
            acc[j] = WMMA_BF16(a, b, acc[j]);
        }
    }
#pragma unroll
    for (int j = 0; j < 4; ++j) {
        int c = col0 + j * 16 + m;
        int head = c / DHEAD, d = c % DHEAD;
        float bi = bias[c];
#pragma unroll
        for (int r = 0; r < 8; ++r) {
            int rr = row0 + h * 8 + r;
            int bb = rr / SEQ, s = rr % SEQ;
            bf16 v = (bf16)(acc[j][r] + bi);
            if (!transposed_out)
                outp[(((size_t)(bb * NHEADS + head)) * SEQ + s) * DHEAD + d] = v;
            else
                outp[(((size_t)(bb * NHEADS + head)) * DHEAD + d) * SEQ + s] = v;
        }
    }
}

// ---------------- scores = q @ k^T * 0.125 ----------------
// One block computes a 128x64 score tile for one head. The shared 64x64 bf16
// K-panel is staged into LDS once per block: via the Tensor Data Mover when
// the toolchain exposes it, else a cooperative vector copy. All 8 waves then
// pull B fragments from LDS (ds_load_b128) while streaming A from global.
__global__ void scores_gemm(const bf16* __restrict__ q, const bf16* __restrict__ k,
                            float* __restrict__ attn) {
    const int nblk = SEQ / 64;  // 32 column tiles
    const int mblk = SEQ / 128; // 16 row blocks
    int bid = blockIdx.x;
    int headid = bid / (mblk * nblk);
    if (headid >= NHB) return;
    int t = bid % (mblk * nblk);
    int bm = t / nblk, bn = t % nblk;
    int wave = threadIdx.x >> 5, lane = threadIdx.x & 31;
    int col0 = bn * 64;
    const bf16* qh = q + (size_t)headid * SEQ * DHEAD;
    const bf16* kh = k + (size_t)headid * SEQ * DHEAD;
    float* ah = attn + (size_t)headid * SEQ * SEQ;

    // Only LDS allocation in this kernel -> LDS offset 0 (used by the D#).
    __shared__ __align__(16) bf16 kpanel[64 * DHEAD]; // 8 KB

#if USE_TDM
    if (wave == 0) {
        unsigned long long gaddr = (unsigned long long)(uintptr_t)(kh + (size_t)col0 * DHEAD);
        // D# group 0 (CDNA5 ISA): count=1, lds_addr=0,
        // global_addr, type=2.
        v4u g0;
        g0.x = 1u;                                   // count=1, user mode
        g0.y = 0u;                                   // lds_addr = 0
        g0.z = (unsigned)(gaddr & 0xffffffffu);      // global_addr[31:0]
        g0.w = (unsigned)((gaddr >> 32) & 0x1ffffffu) | (2u << 30); // addr[56:32] | type=2
        // D# group 1 (§8.4): data_size=1 (2B), tensor_dim0=64, tensor_dim1=SEQ,
        // tile_dim0=64, tile_dim1=64, tensor_dim0_stride=64.
        v8i g1;
        g1[0] = (1 << 16);        // workgroup_mask=0, data_size=1 (2 bytes)
        g1[1] = (64 << 16);       // tensor_dim0[15:0]=64 in bits 63:48
        g1[2] = (SEQ << 16);      // tensor_dim0[31:16]=0, tensor_dim1[15:0]=SEQ
        g1[3] = (64 << 16);       // tensor_dim1[31:16]=0, tile_dim0=64
        g1[4] = 64;               // tile_dim1=64, tile_dim2=0
        g1[5] = 64;               // tensor_dim0_stride[31:0]=64
        g1[6] = 0;                // stride[47:32]=0, tensor_dim1_stride[15:0]=0
        g1[7] = 0;
        v4i gz = {0, 0, 0, 0};
#if __clang_major__ >= 23
        v8i gz8 = {0, 0, 0, 0, 0, 0, 0, 0};
        __builtin_amdgcn_tensor_load_to_lds(g0, g1, gz, gz, gz8, 0);
#else
        __builtin_amdgcn_tensor_load_to_lds(g0, g1, gz, gz, 0);
#endif
        __builtin_amdgcn_s_wait_tensorcnt(0);
    }
#else
    {
        int row = threadIdx.x >> 2, c16 = (threadIdx.x & 3) * 16;
        *(v8bf*)&kpanel[row * DHEAD + c16] = *(const v8bf*)&kh[(size_t)(col0 + row) * DHEAD + c16];
        *(v8bf*)&kpanel[row * DHEAD + c16 + 8] =
            *(const v8bf*)&kh[(size_t)(col0 + row) * DHEAD + c16 + 8];
    }
#endif
    __syncthreads();

    int row0 = bm * 128 + wave * 16;
    int m = lane & 15, h = lane >> 4;
    v8f acc[4] = {{}, {}, {}, {}};
#pragma unroll
    for (int k0 = 0; k0 < DHEAD; k0 += 32) {
        v16bf a = load_frag_bf16(qh + (size_t)(row0 + m) * DHEAD + k0, h);
#pragma unroll
        for (int j = 0; j < 4; ++j) {
            v16bf b = load_frag_bf16(&kpanel[(j * 16 + m) * DHEAD + k0], h);
            acc[j] = WMMA_BF16(a, b, acc[j]);
        }
    }
#pragma unroll
    for (int j = 0; j < 4; ++j)
#pragma unroll
        for (int r = 0; r < 8; ++r)
            ah[(size_t)(row0 + h * 8 + r) * SEQ + col0 + j * 16 + m] = acc[j][r] * 0.125f;
}

// ---------------- row softmax, one wave32 per row of 2048, row in VGPRs ----------------
__global__ void softmax_rows(float* __restrict__ attn) {
    int wid = blockIdx.x * (blockDim.x >> 5) + (threadIdx.x >> 5);
    int lane = threadIdx.x & 31;
    if (wid >= NHB * SEQ) return;
    float* row = attn + (size_t)wid * SEQ;

    float vals[SEQ / 32];
    float mx = -3.402823e38f;
#pragma unroll
    for (int i = 0; i < SEQ / 32; ++i) {
        vals[i] = row[lane + i * 32];
        mx = fmaxf(mx, vals[i]);
    }
#pragma unroll
    for (int off = 16; off > 0; off >>= 1) mx = fmaxf(mx, __shfl_xor(mx, off, 32));
    float sum = 0.0f;
#pragma unroll
    for (int i = 0; i < SEQ / 32; ++i) {
        vals[i] = __expf(vals[i] - mx);
        sum += vals[i];
    }
#pragma unroll
    for (int off = 16; off > 0; off >>= 1) sum += __shfl_xor(sum, off, 32);
    float inv = 1.0f / sum;
#pragma unroll
    for (int i = 0; i < SEQ / 32; ++i) row[lane + i * 32] = vals[i] * inv;
}

// ---------------- context = attn @ v ; vT [NHB,64,S] bf16 -> ctx [B,S,1024] bf16 ----------------
__global__ void context_gemm(const float* __restrict__ attn, const bf16* __restrict__ vT,
                             bf16* __restrict__ ctx) {
    int wid = blockIdx.x * (blockDim.x >> 5) + (threadIdx.x >> 5);
    int lane = threadIdx.x & 31;
    const int tiles_per_head = SEQ / 16; // 128
    int headid = wid / tiles_per_head;
    if (headid >= NHB) return;
    int tm = wid % tiles_per_head;
    const float* ah = attn + (size_t)headid * SEQ * SEQ;
    const bf16* vh = vT + (size_t)headid * DHEAD * SEQ;
    int row0 = tm * 16;
    int m = lane & 15, h = lane >> 4;

    const float* arow = ah + (size_t)(row0 + m) * SEQ;
    v8f acc[4] = {{}, {}, {}, {}};
    for (int k0 = 0; k0 < SEQ; k0 += 32) {
        __builtin_prefetch(arow + k0 + 64, 0, 1);
        v16bf a = load_frag_f32(arow + k0, h);
#pragma unroll
        for (int j = 0; j < 4; ++j) {
            v16bf b = load_frag_bf16(vh + (size_t)(j * 16 + m) * SEQ + k0, h);
            acc[j] = WMMA_BF16(a, b, acc[j]);
        }
    }
    int bb = headid / NHEADS, head = headid % NHEADS;
#pragma unroll
    for (int j = 0; j < 4; ++j)
#pragma unroll
        for (int r = 0; r < 8; ++r) {
            int s = row0 + h * 8 + r;
            ctx[((size_t)(bb * SEQ + s)) * DMODEL + head * DHEAD + j * 16 + m] =
                (bf16)acc[j][r];
        }
}

// ---------------- x = ctx @ Wo + bo + Q (residual); WoT [N,K] bf16 ----------------
__global__ void out_gemm(const bf16* __restrict__ ctx, const bf16* __restrict__ WoT,
                         const float* __restrict__ bo, const float* __restrict__ Qres,
                         float* __restrict__ x) {
    int wid = blockIdx.x * (blockDim.x >> 5) + (threadIdx.x >> 5);
    int lane = threadIdx.x & 31;
    const int tiles_n = DMODEL / 64; // 16
    int tm = wid / tiles_n, tn = wid % tiles_n;
    if (tm >= MTOT / 16) return;
    int row0 = tm * 16, col0 = tn * 64;
    int m = lane & 15, h = lane >> 4;

    const bf16* arow = ctx + (size_t)(row0 + m) * DMODEL;
    v8f acc[4] = {{}, {}, {}, {}};
    for (int k0 = 0; k0 < DMODEL; k0 += 32) {
        __builtin_prefetch(arow + k0 + 64, 0, 1);
        v16bf a = load_frag_bf16(arow + k0, h);
#pragma unroll
        for (int j = 0; j < 4; ++j) {
            v16bf b = load_frag_bf16(WoT + (size_t)(col0 + j * 16 + m) * DMODEL + k0, h);
            acc[j] = WMMA_BF16(a, b, acc[j]);
        }
    }
#pragma unroll
    for (int j = 0; j < 4; ++j) {
        int c = col0 + j * 16 + m;
        float bi = bo[c];
#pragma unroll
        for (int r = 0; r < 8; ++r) {
            int rr = row0 + h * 8 + r;
            x[(size_t)rr * DMODEL + c] = acc[j][r] + bi + Qres[(size_t)rr * DMODEL + c];
        }
    }
}

// ---------------- LayerNorm over last dim, one wave32 per row of 1024 ----------------
__global__ void layernorm_rows(const float* __restrict__ x, const float* __restrict__ g,
                               const float* __restrict__ bta, float* __restrict__ out) {
    int wid = blockIdx.x * (blockDim.x >> 5) + (threadIdx.x >> 5);
    int lane = threadIdx.x & 31;
    if (wid >= MTOT) return;
    const float* r = x + (size_t)wid * DMODEL;

    float vals[DMODEL / 32];
    float sum = 0.0f, sumsq = 0.0f;
#pragma unroll
    for (int i = 0; i < DMODEL / 32; ++i) {
        float v = r[lane + i * 32];
        vals[i] = v;
        sum += v;
        sumsq += v * v;
    }
#pragma unroll
    for (int off = 16; off > 0; off >>= 1) {
        sum += __shfl_xor(sum, off, 32);
        sumsq += __shfl_xor(sumsq, off, 32);
    }
    float mu = sum * (1.0f / DMODEL);
    float var = sumsq * (1.0f / DMODEL) - mu * mu;
    float rstd = rsqrtf(var + 1e-5f);
#pragma unroll
    for (int i = 0; i < DMODEL / 32; ++i) {
        int c = lane + i * 32;
        out[(size_t)wid * DMODEL + c] = (vals[i] - mu) * rstd * g[c] + bta[c];
    }
}

extern "C" void kernel_launch(void* const* d_in, const int* in_sizes, int n_in,
                              void* d_out, int out_size, void* d_ws, size_t ws_size,
                              hipStream_t stream) {
    (void)in_sizes; (void)n_in; (void)out_size; (void)ws_size;
    const float* Q    = (const float*)d_in[0];
    const float* K    = (const float*)d_in[1];
    const float* V    = (const float*)d_in[2];
    const float* Wq   = (const float*)d_in[3];
    const float* bq   = (const float*)d_in[4];
    const float* Wk   = (const float*)d_in[5];
    const float* bk   = (const float*)d_in[6];
    const float* Wv   = (const float*)d_in[7];
    const float* bv   = (const float*)d_in[8];
    const float* Wo   = (const float*)d_in[9];
    const float* bo   = (const float*)d_in[10];
    const float* ln_g = (const float*)d_in[11];
    const float* ln_b = (const float*)d_in[12];

    float* out  = (float*)d_out;                 // [B,S,DMODEL]
    float* attn = out + (size_t)MTOT * DMODEL;   // [B,H,S,S]

    // Workspace carve (~56 MB)
    char* ws = (char*)d_ws;
    bf16* WqT = (bf16*)ws; ws += (size_t)DMODEL * DMODEL * sizeof(bf16);
    bf16* WkT = (bf16*)ws; ws += (size_t)DMODEL * DMODEL * sizeof(bf16);
    bf16* WvT = (bf16*)ws; ws += (size_t)DMODEL * DMODEL * sizeof(bf16);
    bf16* WoT = (bf16*)ws; ws += (size_t)DMODEL * DMODEL * sizeof(bf16);
    bf16* qh  = (bf16*)ws; ws += (size_t)NHB * SEQ * DHEAD * sizeof(bf16);
    bf16* kh  = (bf16*)ws; ws += (size_t)NHB * SEQ * DHEAD * sizeof(bf16);
    bf16* vT  = (bf16*)ws; ws += (size_t)NHB * DHEAD * SEQ * sizeof(bf16);
    bf16* ctx = (bf16*)ws; ws += (size_t)MTOT * DMODEL * sizeof(bf16);
    float* xbuf = (float*)ws;

    const int THR = 256; // 8 waves/block (wave32)
    dim3 tthr(32, 8);
    const int TBLK = (DMODEL / 32) * (DMODEL / 32); // 1024

    transpose_cvt<<<TBLK, tthr, 0, stream>>>(Wq, WqT);
    transpose_cvt<<<TBLK, tthr, 0, stream>>>(Wk, WkT);
    transpose_cvt<<<TBLK, tthr, 0, stream>>>(Wv, WvT);
    transpose_cvt<<<TBLK, tthr, 0, stream>>>(Wo, WoT);

    proj_gemm<<<512, THR, 0, stream>>>(Q, WqT, bq, qh, 0);
    proj_gemm<<<512, THR, 0, stream>>>(K, WkT, bk, kh, 0);
    proj_gemm<<<512, THR, 0, stream>>>(V, WvT, bv, vT, 1);

    // 32 heads * 16 row-blocks * 32 col-tiles = 16384 blocks (8 waves each)
    scores_gemm<<<16384, THR, 0, stream>>>(qh, kh, attn);

    softmax_rows<<<8192, THR, 0, stream>>>(attn);

    context_gemm<<<512, THR, 0, stream>>>(attn, vT, ctx);

    out_gemm<<<512, THR, 0, stream>>>(ctx, WoT, bo, Q, xbuf);

    layernorm_rows<<<512, THR, 0, stream>>>(xbuf, ln_g, ln_b, out);
}
